// BiEvidenceNet_87531433492885
// MI455X (gfx1250) — matrix operand order repaired and checked by reading, not compiled, not run on hardware
//
#include <hip/hip_runtime.h>

// Problem constants (from reference)
#define BB 1024
#define RR 512
#define DD 256
#define RD (RR*DD)
#define BETA_C 6.0f

typedef __attribute__((ext_vector_type(16))) _Float16 v16h;
typedef __attribute__((ext_vector_type(8)))  _Float16 v8h;
typedef __attribute__((ext_vector_type(8)))  float    v8f;

// ---------- fast math helpers (map to v_exp_f32 / v_rcp_f32 / v_tanh_f32) ----------
__device__ __forceinline__ float fexp(float v) {
    return __builtin_amdgcn_exp2f(v * 1.44269504088896340736f);
}
__device__ __forceinline__ float fsig(float u) {
    return __builtin_amdgcn_rcpf(1.0f + __builtin_amdgcn_exp2f(-u * 1.44269504088896340736f));
}
__device__ __forceinline__ float ftanh(float u) {
#if __has_builtin(__builtin_amdgcn_tanhf)
    return __builtin_amdgcn_tanhf(u);        // CDNA5 hardware TANH
#else
    float e = __builtin_amdgcn_exp2f(u * 2.88539008177792681472f); // 2*log2(e)
    return 1.0f - 2.0f * __builtin_amdgcn_rcpf(e + 1.0f);
#endif
}
__device__ __forceinline__ float fclamp(float v, float lo, float hi) {
    return fminf(fmaxf(v, lo), hi);
}

// ---------- CDNA5 async global->LDS copy (ASYNCcnt-tracked DMA) ----------
// GLOBAL_LOAD_ASYNC_TO_LDS_B128: VDST = LDS byte address (low 32 bits of the
// generic shared pointer), VADDR = 64-bit global address.  Per-lane 16B.
__device__ __forceinline__ void async_copy_b128(void* lds_dst, const void* gsrc) {
    unsigned loff = (unsigned)(uintptr_t)lds_dst;
    asm volatile("global_load_async_to_lds_b128 %0, %1, off"
                 :: "v"(loff), "v"(gsrc)
                 : "memory");
}
__device__ __forceinline__ void wait_async0() {
    asm volatile("s_wait_asynccnt 0x0" ::: "memory");
}

// ---------- Kernel A: fold (R,D) params; amortized over all B rows ----------
__global__ __launch_bounds__(256) void prep_kernel(
    const float* __restrict__ center, const float* __restrict__ log_width,
    const float* __restrict__ e_low,  const float* __restrict__ e_high,
    const float* __restrict__ mask,   const float* __restrict__ log_kappa,
    float* __restrict__ P1, float* __restrict__ P2,
    float* __restrict__ HQ1, float* __restrict__ HQ2)
{
    int i = blockIdx.x * 256 + threadIdx.x;
    float kappa = fclamp(fexp(log_kappa[0]), 0.5f, 50.0f);
    float h = 0.5f * kappa;
    float w = fclamp(fexp(log_width[i]), 0.001f, 50.0f);
    float c = center[i];
    float m = fsig(mask[i]);
    P1[i]  = m * ftanh(e_low[i]);
    P2[i]  = m * ftanh(e_high[i]);
    HQ1[i] = h * (c - 0.5f * w);
    HQ2[i] = h * (c + 0.5f * w);
}

// ---------- Kernel B: evidence + z, 16(b) x 16(r) tile per block ----------
// evidence[b,r] = sum_d P1*tanh(HQ1 - h*x) + P2*tanh(h*x - HQ2)
// Double-buffered: async-DMA chunk k+1 into LDS while the VALU computes
// chunk k; the half-kappa scale of x is folded into the inner-loop FMAs.
__global__ __launch_bounds__(256) void evid_kernel(
    const float* __restrict__ x, const float* __restrict__ log_kappa,
    const float* __restrict__ tvec,
    const float* __restrict__ P1, const float* __restrict__ P2,
    const float* __restrict__ HQ1, const float* __restrict__ HQ2,
    _Float16* __restrict__ zh)
{
    // stride 68: 16B-aligned rows, conflict-free reads (bank = (4*row+col)%64)
    __shared__ float xs [2][16][68];
    __shared__ float p1s[2][16][68];
    __shared__ float p2s[2][16][68];
    __shared__ float q1s[2][16][68];
    __shared__ float q2s[2][16][68];

    const int tid = threadIdx.x;
    const int tx  = tid & 15;   // r within tile
    const int ty  = tid >> 4;   // b within tile
    const int b0  = blockIdx.x * 16;
    const int r0  = blockIdx.y * 16;
    const int lrow = tid >> 4;          // staging row (0..15)
    const int lc4  = (tid & 15) * 4;    // staging col (float4 granularity)

    const float h = 0.5f * fclamp(fexp(log_kappa[0]), 0.5f, 50.0f);

    const float* xg  = x   + (size_t)(b0 + lrow) * DD + lc4;
    const float* p1g = P1  + (size_t)(r0 + lrow) * DD + lc4;
    const float* p2g = P2  + (size_t)(r0 + lrow) * DD + lc4;
    const float* q1g = HQ1 + (size_t)(r0 + lrow) * DD + lc4;
    const float* q2g = HQ2 + (size_t)(r0 + lrow) * DD + lc4;

    // prologue: DMA chunk 0 into buffer 0
    async_copy_b128(&xs [0][lrow][lc4], xg);
    async_copy_b128(&p1s[0][lrow][lc4], p1g);
    async_copy_b128(&p2s[0][lrow][lc4], p2g);
    async_copy_b128(&q1s[0][lrow][lc4], q1g);
    async_copy_b128(&q2s[0][lrow][lc4], q2g);
    wait_async0();
    __syncthreads();

    float acc = 0.0f;
    const int NCHUNK = DD / 64;          // 4
    for (int ck = 0; ck < NCHUNK; ++ck) {
        const int cur = ck & 1;
        // issue next chunk into the other buffer (it was last read in
        // iteration ck-1 and is safe to overwrite after that barrier)
        if (ck + 1 < NCHUNK) {
            const int nxt = 1 - cur;
            const int off = (ck + 1) * 64;
            async_copy_b128(&xs [nxt][lrow][lc4], xg  + off);
            async_copy_b128(&p1s[nxt][lrow][lc4], p1g + off);
            async_copy_b128(&p2s[nxt][lrow][lc4], p2g + off);
            async_copy_b128(&q1s[nxt][lrow][lc4], q1g + off);
            async_copy_b128(&q2s[nxt][lrow][lc4], q2g + off);
        }

        #pragma unroll
        for (int dq = 0; dq < 16; ++dq) {
            float4 xv = *(const float4*)&xs [cur][ty][dq * 4];
            float4 a1 = *(const float4*)&p1s[cur][tx][dq * 4];
            float4 a2 = *(const float4*)&p2s[cur][tx][dq * 4];
            float4 b1 = *(const float4*)&q1s[cur][tx][dq * 4];
            float4 b2 = *(const float4*)&q2s[cur][tx][dq * 4];
            // (HQ1 - h*x) and (h*x - HQ2) as single FMAs
            acc += a1.x * ftanh(fmaf(-h, xv.x, b1.x)) + a2.x * ftanh(fmaf(h, xv.x, -b2.x));
            acc += a1.y * ftanh(fmaf(-h, xv.y, b1.y)) + a2.y * ftanh(fmaf(h, xv.y, -b2.y));
            acc += a1.z * ftanh(fmaf(-h, xv.z, b1.z)) + a2.z * ftanh(fmaf(h, xv.z, -b2.z));
            acc += a1.w * ftanh(fmaf(-h, xv.w, b1.w)) + a2.w * ftanh(fmaf(h, xv.w, -b2.w));
        }

        if (ck + 1 < NCHUNK) {
            wait_async0();      // next chunk's DMA done (this wave)
            __syncthreads();    // all waves' DMAs done + all reads of cur done
        }
    }

    float zv = fsig(BETA_C * (acc - tvec[r0 + tx]));
    zh[(size_t)(b0 + ty) * RR + (r0 + tx)] = (_Float16)zv;
}

// ---------- Kernel C: y = z @ head_w^T + head_b via v_wmma_f32_16x16x32_f16 ----------
// One wave per 16 batch rows; head_w replicated across all 16 N columns so
// every output column equals the y tile; lanes 0/16 hold the 16-row result.
__global__ __launch_bounds__(32) void head_kernel(
    const _Float16* __restrict__ zh, const float* __restrict__ head_w,
    const float* __restrict__ head_b, float* __restrict__ y)
{
    const int lane = threadIdx.x;
    const int m = lane & 15;
    const int g = lane >> 4;
    const int b0 = blockIdx.x * 16;
    const _Float16* zrow = zh + (size_t)(b0 + m) * RR;

    v8f c = {};
    #pragma unroll
    for (int k0 = 0; k0 < RR; k0 += 32) {
        v8h lo = *(const v8h*)(zrow + k0 + 8 * g);        // K = k0+8g .. +7
        v8h hi = *(const v8h*)(zrow + k0 + 16 + 8 * g);   // K = k0+16+8g .. +7
        v16h a, b;
        #pragma unroll
        for (int e = 0; e < 8; ++e) { a[e] = lo[e]; a[e + 8] = hi[e]; }
        #pragma unroll
        for (int e = 0; e < 16; ++e) { b[e] = (_Float16)head_w[k0 + 8 * g + (e & 7) + 16 * (e >> 3)]; }
        c = __builtin_amdgcn_wmma_f32_16x16x32_f16(
                /*neg_a=*/false, a, /*neg_b=*/false, b,
                /*c_mod=*/(short)0, c, /*reuse_a=*/false, /*reuse_b=*/false);
    }
    float hb = head_b[0];
    if (lane == 0) {
        #pragma unroll
        for (int i = 0; i < 8; ++i) y[b0 + i] = c[i] + hb;
    }
    if (lane == 16) {
        #pragma unroll
        for (int i = 0; i < 8; ++i) y[b0 + 8 + i] = c[i] + hb;
    }
}

extern "C" void kernel_launch(void* const* d_in, const int* in_sizes, int n_in,
                              void* d_out, int out_size, void* d_ws, size_t ws_size,
                              hipStream_t stream) {
    const float* x         = (const float*)d_in[0];
    const float* center    = (const float*)d_in[1];
    const float* log_width = (const float*)d_in[2];
    const float* e_low     = (const float*)d_in[3];
    const float* e_high    = (const float*)d_in[4];
    const float* mask      = (const float*)d_in[5];
    const float* log_kappa = (const float*)d_in[6];
    const float* tvec      = (const float*)d_in[7];
    const float* head_w    = (const float*)d_in[8];
    const float* head_b    = (const float*)d_in[9];
    float* y = (float*)d_out;

    // workspace layout: 4 x (R,D) f32 param arrays (2 MB) + (B,R) f16 z (1 MB)
    float* P1  = (float*)d_ws;
    float* P2  = P1 + RD;
    float* HQ1 = P2 + RD;
    float* HQ2 = HQ1 + RD;
    _Float16* zh = (_Float16*)(HQ2 + RD);

    prep_kernel<<<RD / 256, 256, 0, stream>>>(center, log_width, e_low, e_high,
                                              mask, log_kappa, P1, P2, HQ1, HQ2);
    evid_kernel<<<dim3(BB / 16, RR / 16), 256, 0, stream>>>(x, log_kappa, tvec,
                                                            P1, P2, HQ1, HQ2, zh);
    head_kernel<<<BB / 16, 32, 0, stream>>>(zh, head_w, head_b, y);
}
